// LiteMLA_44487271252626
// MI455X (gfx1250) — matrix operand order repaired and checked by reading, not compiled, not run on hardware
//
#include <hip/hip_runtime.h>
#include <hip/hip_bf16.h>
#include <stdint.h>

// ---------------------------------------------------------------------------
// LiteMLA fused pipeline for MI455X (gfx1250, wave32, WMMA bf16 16x16x32)
// ---------------------------------------------------------------------------

typedef __attribute__((ext_vector_type(16))) __bf16 v16bf;
typedef __attribute__((ext_vector_type(8)))  float  v8f;

#define TN      16384      // tokens (128*128)
#define WIDTH   128
#define EPS_ATT 1e-15f

__device__ __forceinline__ unsigned short f2bf(float f) {
  unsigned int u = __builtin_bit_cast(unsigned int, f);
  u += 0x7fffu + ((u >> 16) & 1u);          // round-to-nearest-even
  return (unsigned short)(u >> 16);
}
__device__ __forceinline__ float bf2f(unsigned short h) {
  unsigned int u = ((unsigned int)h) << 16;
  return __builtin_bit_cast(float, u);
}

struct B32x2 { uint4 a, b; };
__device__ __forceinline__ v16bf frag2(const unsigned short* p0, const unsigned short* p1) {
  B32x2 t;
  t.a = *(const uint4*)p0;
  t.b = *(const uint4*)p1;
  return __builtin_bit_cast(v16bf, t);
}

// ---------------------------------------------------------------------------
// K0: f32 -> bf16 casts for GEMM operands, fold BN into scale/bias
// ---------------------------------------------------------------------------
__global__ __launch_bounds__(256) void prep_kernel(
    const float* __restrict__ x, const float* __restrict__ wq, const float* __restrict__ pj,
    const float* __restrict__ gam, const float* __restrict__ bet,
    const float* __restrict__ mu,  const float* __restrict__ var,
    unsigned short* __restrict__ xb, unsigned short* __restrict__ wqb,
    unsigned short* __restrict__ pjb, float* __restrict__ bnS, float* __restrict__ bnB)
{
  const int i = blockIdx.x * 256 + threadIdx.x;
  if (i < 256 * TN)   xb[i]  = f2bf(x[i]);
  if (i < 768 * 256)  wqb[i] = f2bf(wq[i]);
  if (i < 256 * 1792) pjb[i] = f2bf(pj[i]);
  if (i < 256) {
    float s = gam[i] * rsqrtf(var[i] + 1e-5f);
    bnS[i] = s;
    bnB[i] = bet[i] - mu[i] * s;
  }
}

// ---------------------------------------------------------------------------
// GEMM: C[M][TN] (f32, optional BN) = A[M][Kd] (bf16) * B[Kd][TN] (bf16)
// Block tile 64(M) x 256(N), K-step 32, 8 waves, each wave 64x32 via
// 4x2 v_wmma_f32_16x16x32_bf16 accumulators.
// ---------------------------------------------------------------------------
__global__ __launch_bounds__(256) void gemm_bf16_wmma(
    const unsigned short* __restrict__ A,
    const unsigned short* __restrict__ B,
    float* __restrict__ C, int Kd,
    const float* __restrict__ bnS, const float* __restrict__ bnB)
{
  __shared__ __align__(16) unsigned short lA[64 * 32];   // [m][k]
  __shared__ __align__(16) unsigned short lB[256 * 32];  // [n][k]
  const int tid  = threadIdx.x;
  const int wv   = tid >> 5;
  const int lane = tid & 31;
  const int hl   = lane & 15;
  const int m0   = blockIdx.y * 64;
  const int n0   = blockIdx.x * 256;

  v8f acc[4][2] = {};

  for (int k0 = 0; k0 < Kd; k0 += 32) {
    for (int i = tid; i < 64 * 32; i += 256)
      lA[i] = A[(m0 + (i >> 5)) * Kd + k0 + (i & 31)];
    for (int i = tid; i < 32 * 256; i += 256) {
      const int k = i >> 8, n = i & 255;
      lB[n * 32 + k] = B[(k0 + k) * TN + n0 + n];
    }
    if (k0 + 32 < Kd) {   // pull next tiles toward L2/L0 (global_prefetch_b8)
      __builtin_prefetch(&B[(size_t)(k0 + 32) * TN + n0 + tid], 0, 1);
      __builtin_prefetch(&A[(size_t)(m0 + (tid & 63)) * Kd + k0 + 32], 0, 1);
    }
    __syncthreads();

    // CDNA5 fragment layouts (ISA 7.12.2):
    //   A 16x32 bf16: lanes 0-15 -> row=hl, K {c0..c0+7, c0+16..c0+23}, c0 = 0|8
    //   B 32x16 bf16: lanes 0-15 -> col=hl, K {c0..c0+7, c0+16..c0+23}, c0 = 0|8
    const int c0 = (lane < 16) ? 0 : 8;
    const int kb = (lane < 16) ? 0 : 16;
    v16bf bfrag[2];
#pragma unroll
    for (int nt = 0; nt < 2; ++nt) {
      const unsigned short* p = &lB[(wv * 32 + nt * 16 + hl) * 32 + c0];
      bfrag[nt] = frag2(p, p + 16);
    }
#pragma unroll
    for (int mt = 0; mt < 4; ++mt) {
      const unsigned short* p = &lA[(mt * 16 + hl) * 32 + c0];
      const v16bf afrag = frag2(p, p + 16);
#pragma unroll
      for (int nt = 0; nt < 2; ++nt) {
        acc[mt][nt] = __builtin_amdgcn_wmma_f32_16x16x32_bf16(
            false, afrag, false, bfrag[nt], (short)0, acc[mt][nt], false, false);
      }
    }
    __syncthreads();
  }

  // D layout: lane 0-15 -> col = hl, VGPR r -> row m0 + r (+8 for lanes 16-31)
  const int mo = (lane < 16) ? 0 : 8;
#pragma unroll
  for (int mt = 0; mt < 4; ++mt) {
#pragma unroll
    for (int nt = 0; nt < 2; ++nt) {
      const int col = n0 + wv * 32 + nt * 16 + hl;
#pragma unroll
      for (int r = 0; r < 8; ++r) {
        const int row = m0 + mt * 16 + mo + r;
        float v = acc[mt][nt][r];
        if (bnS) v = v * bnS[row] + bnB[row];
        C[(size_t)row * TN + col] = v;
      }
    }
  }
}

// ---------------------------------------------------------------------------
// K2: one kernel per "part" (templated). 16x16 spatial tile (+halo 4),
// per head: load 24-channel patch (q,k,v groups), run the part's conv,
// ReLU-q -> bf16 store, accumulate vk = [v;1] k^T via wave32 shfl tree.
// Deterministic: per-(head,tile) partials, summed later in fixed order.
// ---------------------------------------------------------------------------
template <int PART>
__global__ __launch_bounds__(256) void branch_vk_kernel(
    const float* __restrict__ qkv,   // [768][TN] f32
    const float* __restrict__ gw,    // part weights (dw [768][9] or grouped [768][8][9])
    const float* __restrict__ gb,    // bias [768] (parts 3..6)
    const float* __restrict__ gpw,   // grouped 1x1 [768][8]
    unsigned short* __restrict__ qs, // bf16 [1792][TN] (q store, later overwritten)
    float* __restrict__ pvk)         // [224][64][72] partial vk
{
  __shared__ float P[24 * 24 * 24];     // [c][y][x] patch, halo 4
  __shared__ float Wg[24 * 8 * 9];
  __shared__ float Wpw[24 * 8];
  __shared__ float Wb[24];
  __shared__ float wvk[8][72];

  const int tile = blockIdx.x;
  const int ty0 = (tile >> 3) << 4;
  const int tx0 = (tile & 7) << 4;
  const int tid = threadIdx.x;
  const int iy = tid >> 4, ix = tid & 15;
  const int wv = tid >> 5, lane = tid & 31;
  const int py = iy + 4, px = ix + 4;
  const int n = (ty0 + iy) * WIDTH + (tx0 + ix);

  for (int h = 0; h < 32; ++h) {
    const int cbase = h * 24;
    __syncthreads();
    // zero-padded 24ch x 24x24 patch
    for (int i = tid; i < 24 * 24 * 24; i += 256) {
      const int c = i / 576, rem = i - c * 576;
      const int yy = rem / 24, xx = rem - yy * 24;
      const int gy = ty0 + yy - 4, gx = tx0 + xx - 4;
      float v = 0.f;
      if ((unsigned)gy < 128u && (unsigned)gx < 128u)
        v = qkv[(size_t)(cbase + c) * TN + gy * WIDTH + gx];
      P[i] = v;
    }
    if (PART == 1 || PART == 2) {
      for (int i = tid; i < 24 * 9; i += 256) Wg[i] = gw[cbase * 9 + i];
      for (int i = tid; i < 24 * 8; i += 256) Wpw[i] = gpw[cbase * 8 + i];
    } else if (PART >= 3) {
      for (int i = tid; i < 24 * 72; i += 256) Wg[i] = gw[cbase * 72 + i];
      for (int i = tid; i < 24 * 8; i += 256) Wpw[i] = gpw[cbase * 8 + i];
      if (tid < 24) Wb[tid] = gb[cbase + tid];
    }
    __syncthreads();

    float outc[24];
    if (PART == 0) {
#pragma unroll
      for (int c = 0; c < 24; ++c) outc[c] = P[c * 576 + py * 24 + px];
    } else if (PART <= 2) {
      // depthwise 3x3 then grouped 1x1 (8-ch groups)
#pragma unroll
      for (int gi = 0; gi < 3; ++gi) {
        float s[8];
#pragma unroll
        for (int c = 0; c < 8; ++c) s[c] = 0.f;
#pragma unroll
        for (int t = 0; t < 9; ++t) {
          const int oy = t / 3 - 1, ox = t % 3 - 1;
#pragma unroll
          for (int c = 0; c < 8; ++c)
            s[c] += Wg[(gi * 8 + c) * 9 + t] *
                    P[(gi * 8 + c) * 576 + (py + oy) * 24 + (px + ox)];
        }
#pragma unroll
        for (int c = 0; c < 8; ++c) {
          float o = 0.f;
#pragma unroll
          for (int j = 0; j < 8; ++j) o += Wpw[(gi * 8 + c) * 8 + j] * s[j];
          outc[gi * 8 + c] = o;
        }
      }
    } else {
      // grouped 9-tap conv (+bias), then grouped 1x1
#pragma unroll
      for (int gi = 0; gi < 3; ++gi) {
        float s[8];
#pragma unroll
        for (int c = 0; c < 8; ++c) s[c] = Wb[gi * 8 + c];
#pragma unroll
        for (int t = 0; t < 9; ++t) {
          int oy, ox;
          if (PART == 3)      { oy = 0;               ox = t - 4; }        // 1x9
          else if (PART == 4) { oy = t - 4;           ox = 0; }            // 9x1
          else if (PART == 5) { oy = t / 3 - 1;       ox = t % 3 - 1; }    // 3x3
          else                { oy = (t / 3 - 1) * 2; ox = (t % 3 - 1) * 2; } // 3x3 dil2
          float pv[8];
#pragma unroll
          for (int j = 0; j < 8; ++j)
            pv[j] = P[(gi * 8 + j) * 576 + (py + oy) * 24 + (px + ox)];
#pragma unroll
          for (int c = 0; c < 8; ++c)
#pragma unroll
            for (int j = 0; j < 8; ++j)
              s[c] += Wg[((gi * 8 + c) * 8 + j) * 9 + t] * pv[j];
        }
#pragma unroll
        for (int c = 0; c < 8; ++c) {
          float o = 0.f;
#pragma unroll
          for (int j = 0; j < 8; ++j) o += Wpw[(gi * 8 + c) * 8 + j] * s[j];
          outc[gi * 8 + c] = o;
        }
      }
    }

    const int hh = PART * 32 + h;
    float qv[8], kv[8], vv[8];
#pragma unroll
    for (int d = 0; d < 8; ++d) {
      qv[d] = fmaxf(outc[d], 0.f);
      kv[d] = fmaxf(outc[8 + d], 0.f);
      vv[d] = outc[16 + d];
      qs[(size_t)(hh * 8 + d) * TN + n] = f2bf(qv[d]);
    }
    // vk[d][e] partial: rows 0..7 = v_d * k_e, row 8 = sum(k_e)
#pragma unroll
    for (int d = 0; d < 9; ++d) {
      const float vd = (d < 8) ? vv[d] : 1.f;
#pragma unroll
      for (int e = 0; e < 8; ++e) {
        float val = vd * kv[e];
        val += __shfl_down(val, 16, 32);
        val += __shfl_down(val, 8, 32);
        val += __shfl_down(val, 4, 32);
        val += __shfl_down(val, 2, 32);
        val += __shfl_down(val, 1, 32);
        if (lane == 0) wvk[wv][d * 8 + e] = val;
      }
    }
    __syncthreads();
    if (tid < 72) {
      float ssum = 0.f;
#pragma unroll
      for (int w = 0; w < 8; ++w) ssum += wvk[w][tid];
      pvk[(size_t)(hh * 64 + tile) * 72 + tid] = ssum;
    }
  }
}

// ---------------------------------------------------------------------------
// K3: sum per-tile vk partials (fixed order -> deterministic)
// ---------------------------------------------------------------------------
__global__ __launch_bounds__(128) void vk_reduce(const float* __restrict__ pvk,
                                                 float* __restrict__ vk)
{
  const int hh = blockIdx.x;
  const int t = threadIdx.x;
  if (t < 72) {
    float s = 0.f;
    for (int tl = 0; tl < 64; ++tl) s += pvk[(size_t)(hh * 64 + tl) * 72 + t];
    vk[hh * 72 + t] = s;
  }
}

// ---------------------------------------------------------------------------
// K4: out = (vk_num . q) / (vk_den . q + eps), in-place over q-store (bf16)
// ---------------------------------------------------------------------------
__global__ __launch_bounds__(256) void attn_apply(unsigned short* qs,
                                                  const float* __restrict__ vk)
{
  __shared__ float lvk[72];
  const int hh = blockIdx.y;
  const int n = blockIdx.x * 256 + threadIdx.x;
  if (threadIdx.x < 72) lvk[threadIdx.x] = vk[hh * 72 + threadIdx.x];
  __syncthreads();
  float q[8];
#pragma unroll
  for (int e = 0; e < 8; ++e) q[e] = bf2f(qs[(size_t)(hh * 8 + e) * TN + n]);
  float den = EPS_ATT;
#pragma unroll
  for (int e = 0; e < 8; ++e) den += lvk[64 + e] * q[e];
  const float inv = 1.0f / den;
#pragma unroll
  for (int d = 0; d < 8; ++d) {
    float num = 0.f;
#pragma unroll
    for (int e = 0; e < 8; ++e) num += lvk[d * 8 + e] * q[e];
    qs[(size_t)(hh * 8 + d) * TN + n] = f2bf(num * inv);
  }
}

// ---------------------------------------------------------------------------
extern "C" void kernel_launch(void* const* d_in, const int* in_sizes, int n_in,
                              void* d_out, int out_size, void* d_ws, size_t ws_size,
                              hipStream_t stream)
{
  (void)in_sizes; (void)n_in; (void)out_size; (void)ws_size;
  const float* x      = (const float*)d_in[0];
  const float* w_qkv  = (const float*)d_in[1];
  const float* agg_dw = (const float*)d_in[2];
  const float* agg_pw = (const float*)d_in[3];
  const float* brw[4] = {(const float*)d_in[4],  (const float*)d_in[7],
                         (const float*)d_in[10], (const float*)d_in[13]};
  const float* brb[4] = {(const float*)d_in[5],  (const float*)d_in[8],
                         (const float*)d_in[11], (const float*)d_in[14]};
  const float* brp[4] = {(const float*)d_in[6],  (const float*)d_in[9],
                         (const float*)d_in[12], (const float*)d_in[15]};
  const float* proj_w = (const float*)d_in[16];
  const float* gam = (const float*)d_in[17];
  const float* bet = (const float*)d_in[18];
  const float* mu  = (const float*)d_in[19];
  const float* var = (const float*)d_in[20];

  // workspace carve (~123 MB total)
  char* w = (char*)d_ws;
  size_t o = 0;
  auto carve = [&](size_t bytes) -> void* {
    void* p = w + o;
    o += (bytes + 255) & ~(size_t)255;
    return p;
  };
  unsigned short* xb  = (unsigned short*)carve(256ull * TN * 2);     // x bf16
  unsigned short* wqb = (unsigned short*)carve(768ull * 256 * 2);    // w_qkv bf16
  unsigned short* pjb = (unsigned short*)carve(256ull * 1792 * 2);   // proj_w bf16
  float* qkvf = (float*)carve(768ull * TN * 4);                      // qkv f32
  float* pvk  = (float*)carve(224ull * 64 * 72 * 4);                 // vk partials
  float* vk   = (float*)carve(224ull * 72 * 4);                      // reduced vk
  float* bnS  = (float*)carve(256 * 4);
  float* bnB  = (float*)carve(256 * 4);
  unsigned short* qs = (unsigned short*)carve(1792ull * TN * 2);     // q / attn-out bf16

  prep_kernel<<<TN, 256, 0, stream>>>(x, w_qkv, proj_w, gam, bet, mu, var,
                                      xb, wqb, pjb, bnS, bnB);
  // qkv = W(768x256) * x(256xN)
  gemm_bf16_wmma<<<dim3(64, 12), 256, 0, stream>>>(wqb, xb, qkvf, 256, nullptr, nullptr);
  // 7 parts: convs + q-store + vk partials
  branch_vk_kernel<0><<<64, 256, 0, stream>>>(qkvf, nullptr, nullptr, nullptr, qs, pvk);
  branch_vk_kernel<1><<<64, 256, 0, stream>>>(qkvf, agg_dw,            nullptr, agg_pw,           qs, pvk);
  branch_vk_kernel<2><<<64, 256, 0, stream>>>(qkvf, agg_dw + 768 * 9,  nullptr, agg_pw + 768 * 8, qs, pvk);
  branch_vk_kernel<3><<<64, 256, 0, stream>>>(qkvf, brw[0], brb[0], brp[0], qs, pvk);
  branch_vk_kernel<4><<<64, 256, 0, stream>>>(qkvf, brw[1], brb[1], brp[1], qs, pvk);
  branch_vk_kernel<5><<<64, 256, 0, stream>>>(qkvf, brw[2], brb[2], brp[2], qs, pvk);
  branch_vk_kernel<6><<<64, 256, 0, stream>>>(qkvf, brw[3], brb[3], brp[3], qs, pvk);
  vk_reduce<<<224, 128, 0, stream>>>(pvk, vk);
  attn_apply<<<dim3(64, 224), 256, 0, stream>>>(qs, vk);
  // y = BN(P(256x1792) * out(1792xN))
  gemm_bf16_wmma<<<dim3(64, 4), 256, 0, stream>>>(pjb, qs, (float*)d_out, 1792, bnS, bnB);
}
